// FlashMHA_566935683121
// MI455X (gfx1250) — matrix operand order repaired and checked by reading, not compiled, not run on hardware
//
#include <hip/hip_runtime.h>

// ---------------------------------------------------------------------------
// CDNA5 (gfx1250) fused attention block: Q/KV proj + RoPE + GQA flash-attn +
// out proj. Matrix math via v_wmma_f32_16x16x32_bf16 (wave32); GEMM tile
// staging via the Tensor Data Mover (tensor_load_to_lds + s_wait_tensorcnt)
// when the builtin is available, manual LDS staging otherwise.
// Workspace requirement: ~139 MB (see kernel_launch offsets).
// ---------------------------------------------------------------------------

#define DEV static __device__ __forceinline__

typedef __attribute__((ext_vector_type(16))) __bf16 v16bf;
typedef __attribute__((ext_vector_type(8)))  float  v8f;

union V16 { v16bf v; uint4 q[2]; unsigned short u[16]; };
union U2  { uint2 d; unsigned short u[4]; };

static constexpr int Bb  = 8;
static constexpr int S   = 1024;
static constexpr int E   = 2048;
static constexpr int H   = 16;
static constexpr int HKV = 4;
static constexpr int D   = 128;
static constexpr int ROT = 64;
static constexpr int M   = Bb * S;        // 8192 rows
static constexpr int NKV = 2 * HKV * D;   // 1024

DEV unsigned short f2bf(float f) {                // round-to-nearest-even
  unsigned u = __float_as_uint(f);
  u += 0x7FFFu + ((u >> 16) & 1u);
  return (unsigned short)(u >> 16);
}
DEV float bf2f(unsigned short h) { return __uint_as_float(((unsigned)h) << 16); }

// ---------------------------------------------------------------------------
// Tensor Data Mover: DMA a [rows x 32] bf16 tile (row stride = Kk elements)
// from global into LDS with a 40-short padded row pitch (16 data DWORDs +
// 4 pad DWORDs per row, i.e. pad_interval=3, pad_amount=3 per D# encoding).
// ---------------------------------------------------------------------------
#if defined(__has_builtin)
#if __has_builtin(__builtin_amdgcn_tensor_load_to_lds) && \
    __has_builtin(__builtin_amdgcn_s_wait_tensorcnt)
#define USE_TDM 1
#endif
#endif

#ifdef USE_TDM
typedef unsigned int tdm_v4u __attribute__((ext_vector_type(4)));
typedef int          tdm_v4i __attribute__((ext_vector_type(4)));
typedef int          tdm_v8i __attribute__((ext_vector_type(8)));

DEV void tdm_load_tile_32w(const unsigned short* gsrc, unsigned lds_byte,
                           unsigned tensor_rows, unsigned Kk) {
  unsigned long long ga = (unsigned long long)(uintptr_t)gsrc;
  // D# group 0: count=1 | lds_addr | global_addr[56:0] | type=2 (bits 127:126)
  tdm_v4u g0 = { 1u,
                 lds_byte,
                 (unsigned)ga,
                 (unsigned)((ga >> 32) & 0x1FFFFFFull) | (2u << 30) };
  // D# group 1 bitfields (see ISA 8.4):
  //   [17:16] data_size=1 (2B)  [20] pad_enable  [24:22] pad_interval=3 (16 dw)
  //   [31:25] pad_amount=3 (4 dw)  [79:48] tensor_dim0=Kk
  //   [111:80] tensor_dim1=rows   [127:112] tile_dim0=32
  //   [143:128] tile_dim1=128     [207:160] tensor_dim0_stride=Kk
  tdm_v8i g1 = { (int)((1u << 16) | (1u << 20) | (3u << 22) | (3u << 25)),
                 (int)((Kk & 0xFFFFu) << 16),
                 (int)((Kk >> 16) | ((tensor_rows & 0xFFFFu) << 16)),
                 (int)((tensor_rows >> 16) | (32u << 16)),
                 (int)128,
                 (int)Kk,
                 0,
                 0 };
  tdm_v4i gz = { 0, 0, 0, 0 };
#if __clang_major__ >= 23
  tdm_v8i gz8 = { 0, 0, 0, 0, 0, 0, 0, 0 };
  __builtin_amdgcn_tensor_load_to_lds(g0, g1, gz, gz, gz8, 0);
#else
  __builtin_amdgcn_tensor_load_to_lds(g0, g1, gz, gz, 0);
#endif
}
#endif

// ---------------------------------------------------------------------------
__global__ void cvt_f32_bf16(const float* __restrict__ in,
                             unsigned short* __restrict__ out, int n) {
  int i = blockIdx.x * 256 + threadIdx.x;
  if (i < n) out[i] = f2bf(in[i]);
}

// ---------------------------------------------------------------------------
// C[M,N] = A[M,K](bf16) * B[N,K](bf16)^T + bias.  Block: 256 thr (8 waves),
// tile 128x128, waves 2x4, each wave 64x32 via 4x2 WMMA 16x16x32 tiles.
// A/B tiles staged by the TDM (one issuing wave, all waves wait TENSORcnt).
// ---------------------------------------------------------------------------
template <bool OUT_BF16>
__global__ __launch_bounds__(256)
void gemm_bf16_nt(const unsigned short* __restrict__ A,
                  const unsigned short* __restrict__ Bw,
                  const float* __restrict__ bias,
                  void* __restrict__ Cout, int Mm, int Nn, int Kk) {
  __shared__ unsigned short aT[128 * 40];   // 32-wide K slab + 8-short pad
  __shared__ unsigned short bT[128 * 40];

  const int tid  = threadIdx.x;
  const int lane = tid & 31;
  const int wave = tid >> 5;
  const int half = lane >> 4;
  const int l16  = lane & 15;
  const int wm   = wave & 1;    // 0..1  (M direction, 64 rows each)
  const int wn   = wave >> 1;   // 0..3  (N direction, 32 cols each)
  const int mBase = blockIdx.y * 128;
  const int nBase = blockIdx.x * 128;

  v8f acc[4][2] = {};

  for (int k0 = 0; k0 < Kk; k0 += 32) {
#ifdef USE_TDM
    if (wave == 0) {
      tdm_load_tile_32w(&A[(size_t)mBase * Kk + k0],
                        (unsigned)(uintptr_t)&aT[0], (unsigned)Mm,
                        (unsigned)Kk);
      tdm_load_tile_32w(&Bw[(size_t)nBase * Kk + k0],
                        (unsigned)(uintptr_t)&bT[0], (unsigned)Nn,
                        (unsigned)Kk);
    }
    __builtin_amdgcn_s_wait_tensorcnt(0);
    __syncthreads();
#else
#pragma unroll
    for (int i = 0; i < 4; ++i) {                 // A tile 128x32
      int c = tid + 256 * i;
      int row = c >> 3, cq = (c & 7) * 4;
      uint2 dA = *(const uint2*)&A[(size_t)(mBase + row) * Kk + k0 + cq];
      *(uint2*)&aT[row * 40 + cq] = dA;
    }
#pragma unroll
    for (int i = 0; i < 4; ++i) {                 // B tile 128x32
      int c = tid + 256 * i;
      int row = c >> 3, cq = (c & 7) * 4;
      uint2 dB = *(const uint2*)&Bw[(size_t)(nBase + row) * Kk + k0 + cq];
      *(uint2*)&bT[row * 40 + cq] = dB;
    }
    __syncthreads();
#endif

    V16 af[4], bfr[2];
#pragma unroll
    for (int mt = 0; mt < 4; ++mt) {              // A: lane=M row, K=half*8+i(+8)
      int row = wm * 64 + mt * 16 + l16;
      af[mt].q[0] = *(const uint4*)&aT[row * 40 + half * 8];
      af[mt].q[1] = *(const uint4*)&aT[row * 40 + half * 8 + 16];
    }
#pragma unroll
    for (int nt = 0; nt < 2; ++nt) {              // B: lane=N col, K=half*8+i(+8)
      int col = wn * 32 + nt * 16 + l16;
      bfr[nt].q[0] = *(const uint4*)&bT[col * 40 + half * 8];
      bfr[nt].q[1] = *(const uint4*)&bT[col * 40 + half * 8 + 16];
    }
#pragma unroll
    for (int mt = 0; mt < 4; ++mt)
#pragma unroll
      for (int nt = 0; nt < 2; ++nt)
        acc[mt][nt] = __builtin_amdgcn_wmma_f32_16x16x32_bf16(
            false, af[mt].v, false, bfr[nt].v, (short)0, acc[mt][nt], false,
            false);
    __syncthreads();
  }

#pragma unroll
  for (int mt = 0; mt < 4; ++mt)
#pragma unroll
    for (int nt = 0; nt < 2; ++nt)
#pragma unroll
      for (int r = 0; r < 8; ++r) {               // C: row=r+8*half, col=l16
        int gRow = mBase + wm * 64 + mt * 16 + r + 8 * half;
        int gCol = nBase + wn * 32 + nt * 16 + l16;
        float v = acc[mt][nt][r] + bias[gCol];
        if (OUT_BF16)
          ((unsigned short*)Cout)[(size_t)gRow * Nn + gCol] = f2bf(v);
        else
          ((float*)Cout)[(size_t)gRow * Nn + gCol] = v;
      }
}

// ---------------------------------------------------------------------------
// In-place RoPE on bf16 buffer; each thread owns one (d, d+32) pair of one
// head, so the in-place rotate is race-free. rowStride = H*D (q) or 2*HKV*D(k)
// ---------------------------------------------------------------------------
__global__ void rope_kernel(unsigned short* __restrict__ buf, int heads,
                            int rowStride) {
  int idx = blockIdx.x * 256 + threadIdx.x;
  int j = idx & 31;
  int t = idx >> 5;
  int h = t % heads; t /= heads;
  int s = t % S;
  int b = t / S;
  if (b >= Bb) return;
  size_t base = (size_t)(b * S + s) * rowStride + (size_t)h * D;
  float freq = __expf(-(float)j * (2.0f / (float)ROT) * __logf(10000.0f));
  float ang = (float)s * freq;
  float c = __cosf(ang), sn = __sinf(ang);
  float x1 = bf2f(buf[base + j]);
  float x2 = bf2f(buf[base + j + 32]);
  buf[base + j]      = f2bf(x1 * c - x2 * sn);
  buf[base + j + 32] = f2bf(x2 * c + x1 * sn);
}

// ---------------------------------------------------------------------------
// Flash attention, one wave per (b, head, 16-query block). 32-key chunks.
// scores: 8 WMMAs/chunk; P@V: 8 WMMAs/chunk; online softmax with intra-half
// shfl_xor reductions (C-layout rows live inside 16-lane halves on wave32).
// ---------------------------------------------------------------------------
__global__ __launch_bounds__(32)
void flash_attn(const unsigned short* __restrict__ qb,
                const unsigned short* __restrict__ kvb,
                unsigned short* __restrict__ ctx) {
  __shared__ unsigned short kls[32 * 136];   // K chunk [key][d]
  __shared__ unsigned short vls[128 * 40];   // V chunk transposed [d][key]
  __shared__ unsigned short pls[16 * 40];    // P tile [qrow][key]

  const int lane = threadIdx.x;
  const int half = lane >> 4, l16 = lane & 15;
  const int b = blockIdx.z, h = blockIdx.y, q0 = blockIdx.x * 16;
  const int hkv = h >> 2;                    // H/HKV = 4 (GQA)
  const float scale = 0.0883883476f;         // 1/sqrt(128)

  V16 qf[4];
  {
    size_t base = ((size_t)(b * S + q0 + l16) * H + h) * D;
#pragma unroll
    for (int j = 0; j < 4; ++j) {            // A frag: d = 32j + half*8 + i(+8)
      qf[j].q[0] = *(const uint4*)&qb[base + 32 * j + half * 8];
      qf[j].q[1] = *(const uint4*)&qb[base + 32 * j + half * 8 + 16];
    }
  }

  v8f acc[8] = {};
  float mrow[8], lrow[8];
#pragma unroll
  for (int r = 0; r < 8; ++r) { mrow[r] = -1e30f; lrow[r] = 0.f; }

  for (int kk = 0; kk < S; kk += 32) {
    for (int c = lane; c < 1024; c += 32) {  // stage 32 keys x 128 d
      int key = c >> 5, cq = (c & 31) * 4;
      size_t kbase = (((size_t)(b * S + kk + key) * 2 + 0) * HKV + hkv) * D + cq;
      size_t vbase = (((size_t)(b * S + kk + key) * 2 + 1) * HKV + hkv) * D + cq;
      *(uint2*)&kls[key * 136 + cq] = *(const uint2*)&kvb[kbase];
      U2 vv; vv.d = *(const uint2*)&kvb[vbase];
#pragma unroll
      for (int e = 0; e < 4; ++e) vls[(cq + e) * 40 + key] = vv.u[e];
    }
    __syncthreads();

    v8f s0 = {}, s1 = {};
#pragma unroll
    for (int j = 0; j < 4; ++j) {            // B frag: lane=key, k=d=32j+half*8+i(+8)
      V16 kf0, kf1;
      kf0.q[0] = *(const uint4*)&kls[l16 * 136 + 32 * j + half * 8];
      kf0.q[1] = *(const uint4*)&kls[l16 * 136 + 32 * j + half * 8 + 16];
      kf1.q[0] = *(const uint4*)&kls[(16 + l16) * 136 + 32 * j + half * 8];
      kf1.q[1] = *(const uint4*)&kls[(16 + l16) * 136 + 32 * j + half * 8 + 16];
      s0 = __builtin_amdgcn_wmma_f32_16x16x32_bf16(false, qf[j].v, false, kf0.v,
                                                   (short)0, s0, false, false);
      s1 = __builtin_amdgcn_wmma_f32_16x16x32_bf16(false, qf[j].v, false, kf1.v,
                                                   (short)0, s1, false, false);
    }

#pragma unroll
    for (int r = 0; r < 8; ++r) {            // online softmax, row = r + 8*half
      float a0 = s0[r] * scale, a1 = s1[r] * scale;
      float lm = fmaxf(a0, a1);
#pragma unroll
      for (int msk = 1; msk <= 8; msk <<= 1)
        lm = fmaxf(lm, __shfl_xor(lm, msk, 32));
      float mnew = fmaxf(mrow[r], lm);
      float alpha = __expf(mrow[r] - mnew);
      float p0 = __expf(a0 - mnew), p1 = __expf(a1 - mnew);
      float rs = p0 + p1;
#pragma unroll
      for (int msk = 1; msk <= 8; msk <<= 1) rs += __shfl_xor(rs, msk, 32);
      lrow[r] = lrow[r] * alpha + rs;
      mrow[r] = mnew;
#pragma unroll
      for (int n = 0; n < 8; ++n) acc[n][r] *= alpha;
      pls[(r + 8 * half) * 40 + l16]      = f2bf(p0);   // C-layout -> LDS
      pls[(r + 8 * half) * 40 + 16 + l16] = f2bf(p1);
    }
    __syncthreads();

    V16 pf;                                  // P as A frag: lane=qrow, k=key
    pf.q[0] = *(const uint4*)&pls[l16 * 40 + half * 8];
    pf.q[1] = *(const uint4*)&pls[l16 * 40 + half * 8 + 16];
#pragma unroll
    for (int n = 0; n < 8; ++n) {            // V as B frag: lane=d col, k=key
      V16 vf;
      vf.q[0] = *(const uint4*)&vls[(n * 16 + l16) * 40 + half * 8];
      vf.q[1] = *(const uint4*)&vls[(n * 16 + l16) * 40 + half * 8 + 16];
      acc[n] = __builtin_amdgcn_wmma_f32_16x16x32_bf16(false, pf.v, false, vf.v,
                                                       (short)0, acc[n], false,
                                                       false);
    }
    __syncthreads();
  }

#pragma unroll
  for (int n = 0; n < 8; ++n)
#pragma unroll
    for (int r = 0; r < 8; ++r) {
      int srow = q0 + r + 8 * half;
      float v = acc[n][r] / lrow[r];
      ctx[(size_t)(b * S + srow) * E + h * D + n * 16 + l16] = f2bf(v);
    }
}

// ---------------------------------------------------------------------------
extern "C" void kernel_launch(void* const* d_in, const int* in_sizes, int n_in,
                              void* d_out, int out_size, void* d_ws,
                              size_t ws_size, hipStream_t stream) {
  (void)in_sizes; (void)n_in; (void)out_size; (void)ws_size;
  const float* x     = (const float*)d_in[0];
  const float* Wq_w  = (const float*)d_in[1];
  const float* Wq_b  = (const float*)d_in[2];
  const float* Wkv_w = (const float*)d_in[3];
  const float* Wkv_b = (const float*)d_in[4];
  const float* out_w = (const float*)d_in[5];
  const float* out_b = (const float*)d_in[6];
  float* out = (float*)d_out;

  // Workspace layout (bytes); total = 138,412,032 (~132 MiB)
  char* ws = (char*)d_ws;
  size_t o = 0;
  unsigned short* xb  = (unsigned short*)(ws + o); o += (size_t)M * E * 2;
  unsigned short* wq  = (unsigned short*)(ws + o); o += (size_t)E * E * 2;
  unsigned short* wkv = (unsigned short*)(ws + o); o += (size_t)NKV * E * 2;
  unsigned short* wo  = (unsigned short*)(ws + o); o += (size_t)E * E * 2;
  unsigned short* qb  = (unsigned short*)(ws + o); o += (size_t)M * E * 2;
  unsigned short* kvb = (unsigned short*)(ws + o); o += (size_t)M * NKV * 2;
  unsigned short* ctx = (unsigned short*)(ws + o); o += (size_t)M * E * 2;

  cvt_f32_bf16<<<(M * E + 255) / 256, 256, 0, stream>>>(x, xb, M * E);
  cvt_f32_bf16<<<(E * E + 255) / 256, 256, 0, stream>>>(Wq_w, wq, E * E);
  cvt_f32_bf16<<<(NKV * E + 255) / 256, 256, 0, stream>>>(Wkv_w, wkv, NKV * E);
  cvt_f32_bf16<<<(E * E + 255) / 256, 256, 0, stream>>>(out_w, wo, E * E);

  gemm_bf16_nt<true><<<dim3(E / 128, M / 128), 256, 0, stream>>>(
      xb, wq, Wq_b, qb, M, E, E);
  gemm_bf16_nt<true><<<dim3(NKV / 128, M / 128), 256, 0, stream>>>(
      xb, wkv, Wkv_b, kvb, M, NKV, E);

  rope_kernel<<<(Bb * S * H * 32) / 256, 256, 0, stream>>>(qb, H, H * D);
  rope_kernel<<<(Bb * S * HKV * 32) / 256, 256, 0, stream>>>(kvb, HKV,
                                                             2 * HKV * D);

  flash_attn<<<dim3(S / 16, H, Bb), 32, 0, stream>>>(qb, kvb, ctx);

  gemm_bf16_nt<false><<<dim3(E / 128, M / 128), 256, 0, stream>>>(
      ctx, wo, out_b, out, M, E, E);
}
